// GPT2Attention_25520695673280
// MI455X (gfx1250) — compile-verified, matrix operand order checked by reading, not run on hardware
//
#include <hip/hip_runtime.h>

// GPT-2 attention forward for MI455X (gfx1250, wave32, WMMA).
// Bandwidth-bound on the 536 MB fp32 attn_prob output. All GEMMs use
// v_wmma_f32_16x16x32_f16 (f16 in, f32 accumulate). Softmax is fused into the
// score kernel (scores stay VGPR-resident; probs written to HBM exactly once).
// All WMMA operands are loaded so the 8 per-lane f16 pairs form two contiguous
// 16-byte runs -> compiler emits global_load_b128 (verified in round-1 asm).

typedef _Float16 half_t;
typedef __attribute__((ext_vector_type(16))) _Float16 v16h;
typedef __attribute__((ext_vector_type(2)))  _Float16 h2;
typedef __attribute__((ext_vector_type(8)))  float    v8f;
typedef __attribute__((ext_vector_type(2)))  float    f2;

__device__ __forceinline__ v8f wmma16x16x32(v16h a, v16h b, v8f c) {
  // 8 args: (neg_a, A, neg_b, B, c_mod, C, reuse_a, reuse_b)
  return __builtin_amdgcn_wmma_f32_16x16x32_f16(false, a, false, b, (short)0, c,
                                                false, false);
}

// K index pattern for 16-bit A/B tiles (ISA 7.12.2):
// VGPR v in 0..3 -> K = 2v (+8 for lanes 16..31); v in 4..7 -> K = 16+2(v-4) (+8)
__device__ __forceinline__ int kpat(int v, int hsel) {
  return ((v < 4) ? (2 * v) : (16 + 2 * (v - 4))) + 8 * hsel;
}

// A tile 16x32 from f16 row-major, rows m0..m0+15, cols k0..k0+31
__device__ __forceinline__ v16h load_tileA_f16(const half_t* __restrict__ A,
                                               int lda, int m0, int k0, int lane) {
  int hsel = lane >> 4, m = lane & 15;
  const half_t* row = A + (size_t)(m0 + m) * lda + k0;
  v16h r{};
#pragma unroll
  for (int v = 0; v < 8; ++v) {
    int k = kpat(v, hsel);
    h2 p = *(const h2*)(row + k);
    r[2 * v] = p.x;
    r[2 * v + 1] = p.y;
  }
  return r;
}

// A tile 16x32 from f32 row-major, converted to f16 on the fly
__device__ __forceinline__ v16h load_tileA_f32(const float* __restrict__ A,
                                               int lda, int m0, int k0, int lane) {
  int hsel = lane >> 4, m = lane & 15;
  const float* row = A + (size_t)(m0 + m) * lda + k0;
  v16h r{};
#pragma unroll
  for (int v = 0; v < 8; ++v) {
    int k = kpat(v, hsel);
    f2 p = *(const f2*)(row + k);
    r[2 * v] = (_Float16)p.x;
    r[2 * v + 1] = (_Float16)p.y;
  }
  return r;
}

// B tile 32x16 where B = T^T, T row-major NxK (contiguous K pairs -> b128)
__device__ __forceinline__ v16h load_tileB_nk(const half_t* __restrict__ T,
                                              int ldt, int n0, int k0, int lane) {
  int hsel = lane >> 4, n = lane & 15;
  const half_t* row = T + (size_t)(n0 + n) * ldt + k0;
  v16h r{};
#pragma unroll
  for (int v = 0; v < 8; ++v) {
    int k = kpat(v, hsel);
    h2 p = *(const h2*)(row + k);
    r[2 * v] = p.x;
    r[2 * v + 1] = p.y;
  }
  return r;
}

// ---------------- conversion / transpose kernels ----------------

__global__ __launch_bounds__(256) void cvt_f32_f16(const float* __restrict__ in,
                                                   half_t* __restrict__ out, int n) {
  int i = blockIdx.x * blockDim.x + threadIdx.x;
  if (i < n) out[i] = (half_t)in[i];
}

// in: R x C f32 row-major; out: C x R f16 row-major (transposed)
__global__ __launch_bounds__(256) void cvt_transpose_f32_f16(
    const float* __restrict__ in, half_t* __restrict__ out, int R, int C) {
  __shared__ float tile[32][33];
  int c0 = blockIdx.x * 32, r0 = blockIdx.y * 32;
  int tx = threadIdx.x, ty = threadIdx.y;  // (32, 8)
#pragma unroll
  for (int i = ty; i < 32; i += 8)
    tile[i][tx] = in[(size_t)(r0 + i) * C + c0 + tx];
  __syncthreads();
#pragma unroll
  for (int i = ty; i < 32; i += 8)
    out[(size_t)(c0 + i) * R + r0 + tx] = (half_t)tile[tx][i];
}

// Extract V^T per batch from QKVh: vt[b][d][s] = qkvh[b][s][2048 + d]
__global__ __launch_bounds__(256) void transpose_v(const half_t* __restrict__ qkvh,
                                                   half_t* __restrict__ vt) {
  __shared__ half_t tile[32][34];
  int b = blockIdx.z;
  int d0 = blockIdx.x * 32, s0 = blockIdx.y * 32;
  const half_t* src = qkvh + (size_t)b * 2048 * 3072 + 2048;
  half_t* dst = vt + (size_t)b * 1024 * 2048;
  int tx = threadIdx.x, ty = threadIdx.y;  // (32, 8)
#pragma unroll
  for (int i = ty; i < 32; i += 8)
    tile[i][tx] = src[(size_t)(s0 + i) * 3072 + d0 + tx];
  __syncthreads();
#pragma unroll
  for (int i = ty; i < 32; i += 8)
    dst[(size_t)(d0 + i) * 2048 + s0 + tx] = tile[tx][i];
}

// ---------------- GEMM: C[M,N] = A[M,K] @ Bt[N,K]^T + bias ----------------
// Bt is the N x K (transposed) weight -> all operand loads are b128.
// Block = 8 waves; wave owns a 32x32 tile (2x2 WMMA accumulators).
template <bool OUT_F32>
__global__ __launch_bounds__(256) void gemm_rrt(
    const half_t* __restrict__ A, const half_t* __restrict__ Bt,
    const float* __restrict__ bias, void* __restrict__ Cout,
    int K, int lda, int ldc) {
  int lane = threadIdx.x & 31;
  int w = threadIdx.x >> 5;
  int m0 = blockIdx.y * 32;
  int n0 = blockIdx.x * 256 + w * 32;
  v8f acc00 = {}, acc01 = {}, acc10 = {}, acc11 = {};
  for (int k0 = 0; k0 < K; k0 += 32) {
    v16h a0 = load_tileA_f16(A, lda, m0, k0, lane);
    v16h a1 = load_tileA_f16(A, lda, m0 + 16, k0, lane);
    v16h b0 = load_tileB_nk(Bt, K, n0, k0, lane);
    v16h b1 = load_tileB_nk(Bt, K, n0 + 16, k0, lane);
    acc00 = wmma16x16x32(a0, b0, acc00);
    acc01 = wmma16x16x32(a0, b1, acc01);
    acc10 = wmma16x16x32(a1, b0, acc10);
    acc11 = wmma16x16x32(a1, b1, acc11);
  }
  int hsel = lane >> 4, n = lane & 15;
#pragma unroll
  for (int v = 0; v < 8; ++v) {
    int r0 = m0 + v + 8 * hsel;
    int r1 = r0 + 16;
    int c0 = n0 + n, c1 = n0 + 16 + n;
    float x00 = acc00[v] + bias[c0];
    float x01 = acc01[v] + bias[c1];
    float x10 = acc10[v] + bias[c0];
    float x11 = acc11[v] + bias[c1];
    if (OUT_F32) {
      float* C = (float*)Cout;
      C[(size_t)r0 * ldc + c0] = x00;
      C[(size_t)r0 * ldc + c1] = x01;
      C[(size_t)r1 * ldc + c0] = x10;
      C[(size_t)r1 * ldc + c1] = x11;
    } else {
      half_t* C = (half_t*)Cout;
      C[(size_t)r0 * ldc + c0] = (half_t)x00;
      C[(size_t)r0 * ldc + c1] = (half_t)x01;
      C[(size_t)r1 * ldc + c0] = (half_t)x10;
      C[(size_t)r1 * ldc + c1] = (half_t)x11;
    }
  }
}

// ---------------- fused scores + softmax ----------------
// Block = 16 waves (512 threads) owning 16 query rows of one (b,h).
// Wave w holds 16 rows x 128 key-cols in 8 v8f accumulators (64 VGPRs -> no
// spills), applies scale+mask, reduces row max/sum via 16-lane shfl_xor
// butterflies + small LDS cross-wave pass, writes normalized probs once.
__global__ __launch_bounds__(512) void attn_scores_softmax(
    const half_t* __restrict__ qkvh, const float* __restrict__ mask,
    float* __restrict__ attn) {
  const int S = 2048;
  int lane = threadIdx.x & 31;
  int w = threadIdx.x >> 5;  // 0..15 -> key-column chunk of 128
  int bh = blockIdx.y;       // b*16 + h
  int b = bh >> 4, h = bh & 15;
  int m0 = blockIdx.x * 16;

  const half_t* Q = qkvh + (size_t)b * S * 3072 + h * 64;
  const half_t* Km = qkvh + (size_t)b * S * 3072 + 1024 + h * 64;

  v16h aq0 = load_tileA_f16(Q, 3072, m0, 0, lane);
  v16h aq1 = load_tileA_f16(Q, 3072, m0, 32, lane);

  v8f acc[8];
#pragma unroll
  for (int c = 0; c < 8; ++c) {
    int n0 = w * 128 + c * 16;
    v16h bk0 = load_tileB_nk(Km, 3072, n0, 0, lane);   // K^T: contiguous d
    v16h bk1 = load_tileB_nk(Km, 3072, n0, 32, lane);
    v8f t = {};
    t = wmma16x16x32(aq0, bk0, t);
    t = wmma16x16x32(aq1, bk1, t);
    acc[c] = t;
  }

  int hsel = lane >> 4, n = lane & 15;
  const float* Mb = mask + (size_t)b * S * S;  // mask[b,0,:,:]
#pragma unroll
  for (int c = 0; c < 8; ++c) {
    int col = w * 128 + c * 16 + n;
#pragma unroll
    for (int v = 0; v < 8; ++v) {
      int row = m0 + v + 8 * hsel;
      acc[c][v] = acc[c][v] * 0.125f + Mb[(size_t)row * S + col];  // 1/sqrt(64)
    }
  }

  __shared__ float smax[16][16];
  __shared__ float ssum[16][16];

  float pm[8];
#pragma unroll
  for (int v = 0; v < 8; ++v) {
    float m = acc[0][v];
#pragma unroll
    for (int c = 1; c < 8; ++c) m = fmaxf(m, acc[c][v]);
#pragma unroll
    for (int off = 1; off < 16; off <<= 1) m = fmaxf(m, __shfl_xor(m, off, 32));
    pm[v] = m;
  }
  if (n == 0) {
#pragma unroll
    for (int v = 0; v < 8; ++v) smax[w][v + 8 * hsel] = pm[v];
  }
  __syncthreads();

  float gmax[8];
#pragma unroll
  for (int v = 0; v < 8; ++v) {
    int row = v + 8 * hsel;
    float m = smax[0][row];
#pragma unroll
    for (int ww = 1; ww < 16; ++ww) m = fmaxf(m, smax[ww][row]);
    gmax[v] = m;
  }

  float ps[8];
#pragma unroll
  for (int v = 0; v < 8; ++v) {
    float s = 0.f;
#pragma unroll
    for (int c = 0; c < 8; ++c) {
      float e = __expf(acc[c][v] - gmax[v]);
      acc[c][v] = e;
      s += e;
    }
#pragma unroll
    for (int off = 1; off < 16; off <<= 1) s += __shfl_xor(s, off, 32);
    ps[v] = s;
  }
  if (n == 0) {
#pragma unroll
    for (int v = 0; v < 8; ++v) ssum[w][v + 8 * hsel] = ps[v];
  }
  __syncthreads();

  float inv[8];
#pragma unroll
  for (int v = 0; v < 8; ++v) {
    int row = v + 8 * hsel;
    float s = 0.f;
#pragma unroll
    for (int ww = 0; ww < 16; ++ww) s += ssum[ww][row];
    inv[v] = 1.0f / s;
  }

  float* Arow = attn + (size_t)bh * S * S;
#pragma unroll
  for (int c = 0; c < 8; ++c) {
    int col = w * 128 + c * 16 + n;
#pragma unroll
    for (int v = 0; v < 8; ++v) {
      int row = m0 + v + 8 * hsel;
      Arow[(size_t)row * S + col] = acc[c][v] * inv[v];
    }
  }
}

// ---------------- context = P @ V ----------------
// P (fp32 probs) streamed once, converted to f16 on load; V^T gives b128 loads.
// Wave owns a 32x64 tile (full head width) = 8 accumulators.
__global__ __launch_bounds__(256) void attn_context(
    const float* __restrict__ attn, const half_t* __restrict__ vt,
    half_t* __restrict__ ctxh) {
  const int S = 2048;
  int lane = threadIdx.x & 31;
  int w = threadIdx.x >> 5;
  int bh = blockIdx.y;
  int b = bh >> 4, h = bh & 15;
  int m0 = (blockIdx.x * 8 + w) * 32;
  const float* P = attn + (size_t)bh * S * S;
  const half_t* Vt = vt + (size_t)b * 1024 * 2048 + (size_t)(h * 64) * 2048;

  v8f acc[2][4] = {};
  for (int k0 = 0; k0 < S; k0 += 32) {
    v16h a0 = load_tileA_f32(P, S, m0, k0, lane);
    v16h a1 = load_tileA_f32(P, S, m0 + 16, k0, lane);
#pragma unroll
    for (int j = 0; j < 4; ++j) {
      v16h bt = load_tileB_nk(Vt, 2048, j * 16, k0, lane);
      acc[0][j] = wmma16x16x32(a0, bt, acc[0][j]);
      acc[1][j] = wmma16x16x32(a1, bt, acc[1][j]);
    }
  }
  int hsel = lane >> 4, n = lane & 15;
#pragma unroll
  for (int i = 0; i < 2; ++i)
#pragma unroll
    for (int j = 0; j < 4; ++j)
#pragma unroll
      for (int v = 0; v < 8; ++v) {
        int row = m0 + i * 16 + v + 8 * hsel;
        int col = h * 64 + j * 16 + n;
        ctxh[((size_t)b * S + row) * 1024 + col] = (half_t)acc[i][j][v];
      }
}

extern "C" void kernel_launch(void* const* d_in, const int* in_sizes, int n_in,
                              void* d_out, int out_size, void* d_ws, size_t ws_size,
                              hipStream_t stream) {
  const int Bq = 2, S = 2048, D = 1024, H = 16;
  const int MS = Bq * S;  // 4096 total rows

  const float* query = (const float*)d_in[0];
  const float* mask  = (const float*)d_in[1];
  const float* w_qkv = (const float*)d_in[2];
  const float* b_qkv = (const float*)d_in[3];
  const float* w_fc  = (const float*)d_in[4];
  const float* b_fc  = (const float*)d_in[5];

  float* out_ctx  = (float*)d_out;                 // [B,S,D] fp32
  float* out_attn = out_ctx + (size_t)MS * D;      // [B,H,S,S] fp32

  // Workspace layout (f16, ~56 MB total)
  half_t* Xh     = (half_t*)d_ws;                  // query f16   [4096,1024]
  half_t* WqkvT  = Xh + (size_t)MS * D;            // w_qkv^T f16 [3072,1024]
  half_t* WfcT   = WqkvT + (size_t)3 * D * D;      // w_fc^T f16  [1024,1024]
  half_t* QKVh   = WfcT + (size_t)D * D;           // qkv f16     [4096,3072]
  half_t* Ctxh   = QKVh + (size_t)MS * 3 * D;      // ctx f16     [4096,1024]
  half_t* Vt     = Ctxh + (size_t)MS * D;          // V^T f16     [2,1024,2048]

  // casts (+ transposed weights so GEMM B tiles load as b128)
  {
    int n0 = MS * D;
    cvt_f32_f16<<<dim3((n0 + 255) / 256), dim3(256), 0, stream>>>(query, Xh, n0);
    cvt_transpose_f32_f16<<<dim3(3 * D / 32, D / 32), dim3(32, 8), 0, stream>>>(
        w_qkv, WqkvT, D, 3 * D);
    cvt_transpose_f32_f16<<<dim3(D / 32, D / 32), dim3(32, 8), 0, stream>>>(
        w_fc, WfcT, D, D);
  }

  // QKV projection: [4096,3072] = Xh @ WqkvT^T + b_qkv
  gemm_rrt<false><<<dim3(3 * D / 256, MS / 32), dim3(256), 0, stream>>>(
      Xh, WqkvT, b_qkv, (void*)QKVh, D, D, 3 * D);

  // V^T extraction for the context GEMM
  transpose_v<<<dim3(D / 32, S / 32, Bq), dim3(32, 8), 0, stream>>>(QKVh, Vt);

  // Fused scores + softmax -> attn_prob (fp32, written once)
  attn_scores_softmax<<<dim3(S / 16, Bq * H), dim3(512), 0, stream>>>(
      QKVh, mask, out_attn);

  // context = P @ V (P streamed once, f32->f16 on load)
  attn_context<<<dim3(S / 256, Bq * H), dim3(256), 0, stream>>>(
      out_attn, Vt, Ctxh);

  // Output projection: d_out[0:4M) = Ctxh @ WfcT^T + b_fc (fp32)
  gemm_rrt<true><<<dim3(D / 256, MS / 32), dim3(256), 0, stream>>>(
      Ctxh, WfcT, b_fc, d_out, D, D, D);
}
